// BasicMutualTransformerBlock_81527069212979
// MI455X (gfx1250) — matrix operand order repaired and checked by reading, not compiled, not run on hardware
//
#include <hip/hip_runtime.h>
#include <hip/hip_bf16.h>

typedef __bf16 bf16_t;
typedef __attribute__((ext_vector_type(16))) __bf16 v16bf;
typedef __attribute__((ext_vector_type(8)))  float  v8f;
typedef __attribute__((ext_vector_type(4)))  unsigned int u32x4;
typedef __attribute__((ext_vector_type(8)))  int i32x8;
typedef __attribute__((ext_vector_type(4)))  int i32x4;

#define DIMX   512
#define NTOK   2048
#define NHEAD  8
#define DHD    64
#define EPSF   1e-5f
#define SM_SCALE 0.125f   // 64^-0.5

#if defined(__has_builtin)
#  if __has_builtin(__builtin_amdgcn_tensor_load_to_lds)
#    define HAVE_TDM 1
#  endif
#endif
#ifndef HAVE_TDM
#  define HAVE_TDM 0
#endif

// ---------- helpers ----------
__device__ inline bf16_t to_bf(float f) { return (bf16_t)f; }        // RNE fptrunc
__device__ inline unsigned pack2bf(float a, float b) {               // lo=a, hi=b
  union { bf16_t h[2]; unsigned u; } o;
  o.h[0] = (bf16_t)a; o.h[1] = (bf16_t)b; return o.u;
}
// 16B-aligned fragment load (two b128)
__device__ inline v16bf ld_frag16(const bf16_t* p0, const bf16_t* p1) {
  union { uint4 u[2]; v16bf v; } r;
  r.u[0] = *(const uint4*)p0;
  r.u[1] = *(const uint4*)p1;
  return r.v;
}
// 8B-aligned fragment load (four b64)
__device__ inline v16bf ld_frag8(const bf16_t* p0, const bf16_t* p1) {
  union { uint2 u[4]; v16bf v; } r;
  r.u[0] = *(const uint2*)p0;  r.u[1] = *(const uint2*)(p0 + 4);
  r.u[2] = *(const uint2*)p1;  r.u[3] = *(const uint2*)(p1 + 4);
  return r.v;
}

#define WMMA_BF16(A, B, C) \
  __builtin_amdgcn_wmma_f32_16x16x32_bf16(false, (A), false, (B), (short)0, (C), false, false)

// ---------- LayerNorm (f32 in -> bf16 out) ----------
__global__ void __launch_bounds__(128) ln_kernel(const float* __restrict__ x,
                                                 const float* __restrict__ g,
                                                 const float* __restrict__ b,
                                                 bf16_t* __restrict__ out) {
  __shared__ float red[128];
  const int row = blockIdx.x, tid = threadIdx.x;
  const float4 v = ((const float4*)(x + (size_t)row * DIMX))[tid];
  red[tid] = v.x + v.y + v.z + v.w;
  __syncthreads();
  for (int st = 64; st > 0; st >>= 1) { if (tid < st) red[tid] += red[tid + st]; __syncthreads(); }
  const float mu = red[0] * (1.0f / DIMX);
  __syncthreads();
  const float d0 = v.x - mu, d1 = v.y - mu, d2 = v.z - mu, d3 = v.w - mu;
  red[tid] = d0 * d0 + d1 * d1 + d2 * d2 + d3 * d3;
  __syncthreads();
  for (int st = 64; st > 0; st >>= 1) { if (tid < st) red[tid] += red[tid + st]; __syncthreads(); }
  const float rs = rsqrtf(red[0] * (1.0f / DIMX) + EPSF);
  const float4 gg = ((const float4*)g)[tid];
  const float4 bb = ((const float4*)b)[tid];
  union { bf16_t h[4]; uint2 u; } o;
  o.h[0] = to_bf(d0 * rs * gg.x + bb.x);
  o.h[1] = to_bf(d1 * rs * gg.y + bb.y);
  o.h[2] = to_bf(d2 * rs * gg.z + bb.z);
  o.h[3] = to_bf(d3 * rs * gg.w + bb.w);
  *(uint2*)(out + (size_t)row * DIMX + tid * 4) = o.u;
}

// ---------- GEMM: C[M,N] = A(bf16)[M,K] x W(f32->bf16)[K,N] (+bias)(+resid) ----------
// Block tile 64x128, BK=32, 8 waves (wave tile 16x64), double-buffered LDS.
#define SB_STR 44   // sB row stride in halves (22 dw -> conflict-free packed stores)
__global__ void __launch_bounds__(256) gemm_kernel(const bf16_t* __restrict__ A,
                                                   const float* __restrict__ W,
                                                   const float* __restrict__ bias,
                                                   const float* __restrict__ resid,
                                                   void* __restrict__ out,
                                                   int M, int N, int K, int outBf16) {
  __shared__ bf16_t sA[2][64 * 40];        // [m][k] stride 40
  __shared__ bf16_t sB[2][128 * SB_STR];   // [n][k] transposed, stride 44

  const int tid = threadIdx.x;
  const int lane = tid & 31, wave = tid >> 5;
  const int hi = lane >> 4, cn = lane & 15;
  const int wm = wave & 3, wn = wave >> 2;
  const int row0 = blockIdx.y * 64;
  const int col0 = blockIdx.x * 128;

  v8f acc[4];
#pragma unroll
  for (int i = 0; i < 4; i++)
#pragma unroll
    for (int r = 0; r < 8; r++) acc[i][r] = 0.f;

  const int la_r = tid >> 2, la_c = (tid & 3) * 8;      // A: 64 rows x 32 halves
  const int lb_k0 = (tid >> 4) * 2, lb_n = (tid & 15) * 8;  // W: 2 k-rows x 8 cols per thread
  const int nK = K >> 5;

  uint4 areg;
  float4 w0, w1, w2, w3;
  auto ldt = [&](int kt) {   // global -> regs
    const int k0 = kt * 32;
    areg = *(const uint4*)(A + (size_t)(row0 + la_r) * K + k0 + la_c);
    const float* wr0 = W + (size_t)(k0 + lb_k0) * N + col0 + lb_n;
    const float* wr1 = wr0 + N;
    w0 = ((const float4*)wr0)[0]; w1 = ((const float4*)wr0)[1];
    w2 = ((const float4*)wr1)[0]; w3 = ((const float4*)wr1)[1];
  };
  auto stt = [&](int buf) {  // regs -> LDS (convert + transpose, packed u32 stores)
    *(uint4*)&sA[buf][la_r * 40 + la_c] = areg;
    const float a0[8] = {w0.x, w0.y, w0.z, w0.w, w1.x, w1.y, w1.z, w1.w};
    const float b0[8] = {w2.x, w2.y, w2.z, w2.w, w3.x, w3.y, w3.z, w3.w};
#pragma unroll
    for (int i = 0; i < 8; i++)
      *(unsigned*)&sB[buf][(lb_n + i) * SB_STR + lb_k0] = pack2bf(a0[i], b0[i]);
  };

  ldt(0); stt(0); __syncthreads();

  for (int kt = 0; kt < nK; kt++) {
    const int cur = kt & 1;
    if (kt + 1 < nK) ldt(kt + 1);
    if (kt + 2 < nK)
      __builtin_prefetch(W + (size_t)((kt + 2) * 32 + lb_k0) * N + col0 + lb_n, 0, 1);

    const int ar = (wm * 16 + cn) * 40;
    const v16bf af = ld_frag16(&sA[cur][ar + hi * 16], &sA[cur][ar + hi * 16 + 8]);
#pragma unroll
    for (int nt = 0; nt < 4; nt++) {
      const bf16_t* bp = &sB[cur][(wn * 64 + nt * 16 + cn) * SB_STR + hi * 16];
      acc[nt] = WMMA_BF16(af, ld_frag8(bp, bp + 8), acc[nt]);
    }
    if (kt + 1 < nK) stt(cur ^ 1);
    __syncthreads();
  }

  const int rr0 = row0 + wm * 16;
#pragma unroll
  for (int nt = 0; nt < 4; nt++) {
    const int col = col0 + wn * 64 + nt * 16 + cn;
    const float bsv = bias ? bias[col] : 0.f;
#pragma unroll
    for (int r = 0; r < 8; r++) {
      const int row = rr0 + r + 8 * hi;
      float v = acc[nt][r] + bsv;
      if (resid) v += resid[(size_t)row * N + col];
      if (outBf16) ((bf16_t*)out)[(size_t)row * N + col] = to_bf(v);
      else         ((float*)out)[(size_t)row * N + col] = v;
    }
  }
}

// ---------- Flash attention: block = 128 queries x 1 head, 8 waves x 16 rows ----------
#define SQK_STR 72   // sQ/sK row stride in halves (64 data + 8 pad; 144B, conflict-free)
#define SVP_STR 44   // sVT/sP row stride in halves (32 data + 12 pad)
__global__ void __launch_bounds__(256) attn_kernel(const bf16_t* __restrict__ Q,
                                                   const bf16_t* __restrict__ Kp,
                                                   const bf16_t* __restrict__ V,
                                                   bf16_t* __restrict__ O, int Lk) {
  __shared__ bf16_t sQ[128 * SQK_STR];   // [qrow][dh]
  __shared__ bf16_t sK[32 * SQK_STR];    // [key][dh]   (B-layout source for Q.K^T)
  __shared__ bf16_t sVT[64 * SVP_STR];   // [dh][key]   (B-layout source for P.V)
  __shared__ float  sS[8][16 * 33];      // per-wave score scratch 16x32
  __shared__ bf16_t sP[8][16 * SVP_STR]; // per-wave P tile 16x32 (A-layout source)
  __shared__ float  sC[8][16];           // per-wave softmax correction / final l

  const int tid = threadIdx.x, lane = tid & 31, wave = tid >> 5;
  const int hi = lane >> 4, cn = lane & 15;
  const int h = blockIdx.y;
  const int q0 = blockIdx.x * 128;
  const size_t cbase = (size_t)h * DHD;

  { // stage Q: 128 rows x 32 halves (dh 0-15, 32-47); 2 threads/row, 2x16B each
    const int r = tid >> 1, sg = (tid & 1) * 32;
    const uint4* src = (const uint4*)(Q + (size_t)(q0 + r) * DIMX + cbase + sg);
    *(uint4*)&sQ[r * SQK_STR + sg]     = src[0];
    *(uint4*)&sQ[r * SQK_STR + sg + 8] = src[1];
  }
  __syncthreads();

  const int qr = (wave * 16 + cn) * SQK_STR;
  const v16bf aq0 = ld_frag16(&sQ[qr + hi * 16],      &sQ[qr + 8 + hi * 16]);
  const v16bf aq1 = ld_frag16(&sQ[qr + 32 + hi * 16], &sQ[qr + 40 + hi * 16]);

  v8f o0, o1, o2, o3;
#pragma unroll
  for (int r = 0; r < 8; r++) { o0[r] = 0.f; o1[r] = 0.f; o2[r] = 0.f; o3[r] = 0.f; }
  float m_run = -3.0e38f, l_run = 0.f;

  const int vkp = (tid >> 4) * 2, vd4 = (tid & 15) * 4;  // V staging: 2 keys x 4 dh
  const int nKb = Lk / 32;

  for (int kb = 0; kb < nKb; kb++) {
    // ---- stage K tile: Tensor Data Mover (wave 0 issues; others skip) ----
#if HAVE_TDM
    if (wave == 0) {
      const unsigned long long ga =
          (unsigned long long)(uintptr_t)Kp + 2ull * ((size_t)kb * 32 * DIMX + cbase);
      u32x4 g0; i32x8 g1;
      g0[0] = 1u;                                     // count=1, user, no gather
      g0[1] = (unsigned)(uintptr_t)(&sK[0]);          // lds_addr (low 32 = LDS offset)
      g0[2] = (unsigned)ga;                           // global_addr[31:0]
      g0[3] = ((unsigned)(ga >> 32) & 0x01FFFFFFu) | 0x80000000u;  // addr[56:32] | type=2
      g1[0] = (int)((1u << 16) |                      // data_size = 2 bytes
                    (1u << 20) |                      // pad_enable
                    (4u << 22) |                      // pad_interval: 32 dwords (=128B row)
                    (3u << 25));                      // pad_amount: 4 dwords (=16B pad)
      g1[1] = (int)(512u << 16);                      // tensor_dim0 = 512 (low16 in [31:16])
      g1[2] = (int)((unsigned)Lk << 16);              // tensor_dim1 low16
      g1[3] = (int)(64u << 16) | (int)((unsigned)Lk >> 16);  // tile_dim0=64 | dim1 hi16
      g1[4] = 32;                                     // tile_dim1 = 32 keys (tile_dim2=0)
      g1[5] = 512;                                    // tensor_dim0_stride low32
      g1[6] = (int)(512u << 16);                      // dim1_stride low16 (unused for 2D)
      g1[7] = 0;
      const i32x4 z4 = {0, 0, 0, 0};
#if __clang_major__ >= 23
      const i32x8 z8 = {0, 0, 0, 0, 0, 0, 0, 0};
      __builtin_amdgcn_tensor_load_to_lds(g0, g1, z4, z4, z8, 0);
#else
      __builtin_amdgcn_tensor_load_to_lds(g0, g1, z4, z4, 0);
#endif
    }
#else
    { // fallback: vector staging of K
      const int lr = tid >> 3, ls8 = (tid & 7) * 8;
      *(uint4*)&sK[lr * SQK_STR + ls8] =
          *(const uint4*)(Kp + (size_t)(kb * 32 + lr) * DIMX + cbase + ls8);
    }
#endif
    // ---- stage V transposed [dh][key] with packed u32 stores ----
    {
      const bf16_t* vr = V + (size_t)(kb * 32 + vkp) * DIMX + cbase + vd4;
      const uint2 va = *(const uint2*)vr;           // key vkp,   dh vd4..vd4+3
      const uint2 vb = *(const uint2*)(vr + DIMX);  // key vkp+1, dh vd4..vd4+3
      unsigned p0 = (va.x & 0xFFFFu) | (vb.x << 16);
      unsigned p1 = (va.x >> 16)     | (vb.x & 0xFFFF0000u);
      unsigned p2 = (va.y & 0xFFFFu) | (vb.y << 16);
      unsigned p3 = (va.y >> 16)     | (vb.y & 0xFFFF0000u);
      *(unsigned*)&sVT[(vd4 + 0) * SVP_STR + vkp] = p0;
      *(unsigned*)&sVT[(vd4 + 1) * SVP_STR + vkp] = p1;
      *(unsigned*)&sVT[(vd4 + 2) * SVP_STR + vkp] = p2;
      *(unsigned*)&sVT[(vd4 + 3) * SVP_STR + vkp] = p3;
    }
#if HAVE_TDM
    if (wave == 0) __builtin_amdgcn_s_wait_tensorcnt(0);
#endif
    __syncthreads();

    // ---- S = Q K^T (16 q x 32 keys); K-dim = dh = 64 -> 2 wmma per key-subtile ----
    v8f s0, s1;
#pragma unroll
    for (int r = 0; r < 8; r++) { s0[r] = 0.f; s1[r] = 0.f; }
    {
      const bf16_t* b00 = &sK[cn * SQK_STR + hi * 16];
      const bf16_t* b01 = &sK[(16 + cn) * SQK_STR + hi * 16];
      s0 = WMMA_BF16(aq0, ld_frag16(b00, b00 + 8), s0);
      s1 = WMMA_BF16(aq0, ld_frag16(b01, b01 + 8), s1);
      s0 = WMMA_BF16(aq1, ld_frag16(b00 + 32, b00 + 40), s0);
      s1 = WMMA_BF16(aq1, ld_frag16(b01 + 32, b01 + 40), s1);
    }
#pragma unroll
    for (int r = 0; r < 8; r++) {
      sS[wave][(r + 8 * hi) * 33 + cn]      = s0[r];
      sS[wave][(r + 8 * hi) * 33 + 16 + cn] = s1[r];
    }
    // ---- online softmax: lanes 0..15 each own one query row ----
    if (lane < 16) {
      float sv[32]; float mx = -3.0e38f;
#pragma unroll
      for (int c = 0; c < 32; c++) { sv[c] = sS[wave][lane * 33 + c] * SM_SCALE; mx = fmaxf(mx, sv[c]); }
      const float mnew = fmaxf(m_run, mx);
      const float corr = __expf(m_run - mnew);
      float sum = 0.f;
#pragma unroll
      for (int c = 0; c < 32; c++) {
        const float p = __expf(sv[c] - mnew);
        sum += p;
        sP[wave][lane * SVP_STR + c] = to_bf(p);
      }
      l_run = l_run * corr + sum;
      m_run = mnew;
      sC[wave][lane] = corr;
    }
#pragma unroll
    for (int r = 0; r < 8; r++) {
      const float cf = sC[wave][r + 8 * hi];
      o0[r] *= cf; o1[r] *= cf; o2[r] *= cf; o3[r] *= cf;
    }
    // ---- O += P V : A = P(16x32 keys), B = V^T sub-tiles ----
    const v16bf ap = ld_frag8(&sP[wave][cn * SVP_STR + hi * 16],
                              &sP[wave][cn * SVP_STR + 8 + hi * 16]);
    {
      const bf16_t* p;
      p = &sVT[(cn) * SVP_STR + hi * 16];       o0 = WMMA_BF16(ap, ld_frag8(p, p + 8), o0);
      p = &sVT[(16 + cn) * SVP_STR + hi * 16];  o1 = WMMA_BF16(ap, ld_frag8(p, p + 8), o1);
      p = &sVT[(32 + cn) * SVP_STR + hi * 16];  o2 = WMMA_BF16(ap, ld_frag8(p, p + 8), o2);
      p = &sVT[(48 + cn) * SVP_STR + hi * 16];  o3 = WMMA_BF16(ap, ld_frag8(p, p + 8), o3);
    }
    __syncthreads();
  }

  if (lane < 16) sC[wave][lane] = l_run;   // wave-local broadcast of l
#pragma unroll
  for (int r = 0; r < 8; r++) {
    const float inv = 1.f / sC[wave][r + 8 * hi];
    const size_t row = (size_t)(q0 + wave * 16 + r + 8 * hi);
    O[row * DIMX + cbase + cn]      = to_bf(o0[r] * inv);
    O[row * DIMX + cbase + 16 + cn] = to_bf(o1[r] * inv);
    O[row * DIMX + cbase + 32 + cn] = to_bf(o2[r] * inv);
    O[row * DIMX + cbase + 48 + cn] = to_bf(o3[r] * inv);
  }
}

// ---------- GEGLU gate: G[n, c] = H[n, c] * gelu(H[n, 2048+c]) ----------
__global__ void __launch_bounds__(256) geglu_kernel(const float* __restrict__ H,
                                                    bf16_t* __restrict__ G) {
  const int idx = blockIdx.x * 256 + threadIdx.x;
  const int row = idx >> 11, col = idx & 2047;
  const float a  = H[(size_t)row * 4096 + col];
  const float gt = H[(size_t)row * 4096 + 2048 + col];
  const float gl = 0.5f * gt * (1.f + erff(gt * 0.70710678118f));
  G[idx] = to_bf(a * gl);
}

// ---------- host ----------
extern "C" void kernel_launch(void* const* d_in, const int* in_sizes, int n_in,
                              void* d_out, int out_size, void* d_ws, size_t ws_size,
                              hipStream_t stream) {
  (void)in_sizes; (void)n_in; (void)out_size; (void)ws_size;
  const float* x1 = (const float*)d_in[0];
  const float* x2 = (const float*)d_in[1];
  struct MP { const float *q1, *k1, *v1, *q2, *k2, *v2, *ow, *ob; };
  struct AP { const float *q, *k, *v, *ow, *ob; };
  struct FP { const float *w1, *b1, *w2, *b2; };
  const MP m1 = {(const float*)d_in[2], (const float*)d_in[3], (const float*)d_in[4],
                 (const float*)d_in[5], (const float*)d_in[6], (const float*)d_in[7],
                 (const float*)d_in[8], (const float*)d_in[9]};
  const MP m2 = {(const float*)d_in[10], (const float*)d_in[11], (const float*)d_in[12],
                 (const float*)d_in[13], (const float*)d_in[14], (const float*)d_in[15],
                 (const float*)d_in[16], (const float*)d_in[17]};
  const AP a1 = {(const float*)d_in[18], (const float*)d_in[19], (const float*)d_in[20],
                 (const float*)d_in[21], (const float*)d_in[22]};
  const AP a2 = {(const float*)d_in[23], (const float*)d_in[24], (const float*)d_in[25],
                 (const float*)d_in[26], (const float*)d_in[27]};
  const FP mff = {(const float*)d_in[28], (const float*)d_in[29],
                  (const float*)d_in[30], (const float*)d_in[31]};
  const FP ffp = {(const float*)d_in[32], (const float*)d_in[33],
                  (const float*)d_in[34], (const float*)d_in[35]};
  const float* n1g = (const float*)d_in[36]; const float* n1b = (const float*)d_in[37];
  const float* n2g = (const float*)d_in[38]; const float* n2b = (const float*)d_in[39];
  const float* n3g = (const float*)d_in[40]; const float* n3b = (const float*)d_in[41];

  char* ws = (char*)d_ws;
  size_t off = 0;
  auto alloc = [&](size_t bytes) { void* p = ws + off; off = (off + bytes + 255) & ~(size_t)255; return p; };
  const size_t NT = (size_t)NTOK * DIMX;
  bf16_t* A1 = (bf16_t*)alloc(NT * 2);
  bf16_t* A2 = (bf16_t*)alloc(NT * 2);
  bf16_t* Qb = (bf16_t*)alloc(NT * 2);
  bf16_t* Kb = (bf16_t*)alloc(2 * NT * 2);
  bf16_t* Vb = (bf16_t*)alloc(2 * NT * 2);
  bf16_t* Ob = (bf16_t*)alloc(NT * 2);
  float*  X1 = (float*)alloc(NT * 4);
  float*  X2 = (float*)alloc(NT * 4);
  float*  Hb = (float*)alloc((size_t)NTOK * 4096 * 4);
  bf16_t* Gb = (bf16_t*)alloc((size_t)NTOK * 2048 * 2);

  float* out_x1 = (float*)d_out;
  float* out_x2 = (float*)d_out + NT;

  auto LN = [&](const float* x, const float* g, const float* b, bf16_t* o) {
    ln_kernel<<<NTOK, 128, 0, stream>>>(x, g, b, o);
  };
  auto GEMM = [&](const bf16_t* A, const float* W, const float* bias, const float* resid,
                  void* o, int M, int N, int Kd, int obf) {
    gemm_kernel<<<dim3(N / 128, M / 64), 256, 0, stream>>>(A, W, bias, resid, o, M, N, Kd, obf);
  };
  auto ATTN = [&](int Lk) {
    attn_kernel<<<dim3(NTOK / 128, NHEAD), 256, 0, stream>>>(Qb, Kb, Vb, Ob, Lk);
  };

  // ---- layer 1 ----
  LN(x1, n1g, n1b, A1); LN(x2, n1g, n1b, A2);
  GEMM(A1, m1.q1, nullptr, nullptr, Qb, NTOK, 512, 512, 1);
  GEMM(A1, m1.k1, nullptr, nullptr, Kb, NTOK, 512, 512, 1);
  GEMM(A2, m1.k2, nullptr, nullptr, Kb + NT, NTOK, 512, 512, 1);
  GEMM(A1, m1.v1, nullptr, nullptr, Vb, NTOK, 512, 512, 1);
  GEMM(A2, m1.v2, nullptr, nullptr, Vb + NT, NTOK, 512, 512, 1);
  ATTN(4096);
  GEMM(Ob, m1.ow, m1.ob, x1, X1, NTOK, 512, 512, 0);
  GEMM(A2, a1.q, nullptr, nullptr, Qb, NTOK, 512, 512, 1);
  GEMM(A2, a1.k, nullptr, nullptr, Kb, NTOK, 512, 512, 1);
  GEMM(A2, a1.v, nullptr, nullptr, Vb, NTOK, 512, 512, 1);
  ATTN(2048);
  GEMM(Ob, a1.ow, a1.ob, x2, X2, NTOK, 512, 512, 0);

  // ---- layer 2 ----
  LN(X1, n2g, n2b, A1); LN(X2, n2g, n2b, A2);
  GEMM(A1, m2.q1, nullptr, nullptr, Qb, NTOK, 512, 512, 1);
  GEMM(A1, m2.k1, nullptr, nullptr, Kb, NTOK, 512, 512, 1);
  GEMM(A2, m2.k2, nullptr, nullptr, Kb + NT, NTOK, 512, 512, 1);
  GEMM(A1, m2.v1, nullptr, nullptr, Vb, NTOK, 512, 512, 1);
  GEMM(A2, m2.v2, nullptr, nullptr, Vb + NT, NTOK, 512, 512, 1);
  ATTN(4096);
  GEMM(Ob, m2.ow, m2.ob, X1, X1, NTOK, 512, 512, 0);
  GEMM(A2, a2.q, nullptr, nullptr, Qb, NTOK, 512, 512, 1);
  GEMM(A2, a2.k, nullptr, nullptr, Kb, NTOK, 512, 512, 1);
  GEMM(A2, a2.v, nullptr, nullptr, Vb, NTOK, 512, 512, 1);
  ATTN(2048);
  GEMM(Ob, a2.ow, a2.ob, X2, X2, NTOK, 512, 512, 0);

  // ---- feed-forward ----
  LN(X1, n3g, n3b, A1);
  GEMM(A1, mff.w1, mff.b1, nullptr, Hb, NTOK, 4096, 512, 0);
  geglu_kernel<<<(NTOK * 2048) / 256, 256, 0, stream>>>(Hb, Gb);
  GEMM(Gb, mff.w2, mff.b2, X1, out_x1, NTOK, 512, 2048, 0);

  LN(X2, n3g, n3b, A2);
  GEMM(A2, ffp.w1, ffp.b1, nullptr, Hb, NTOK, 4096, 512, 0);
  geglu_kernel<<<(NTOK * 2048) / 256, 256, 0, stream>>>(Hb, Gb);
  GEMM(Gb, ffp.w2, ffp.b2, X2, out_x2, NTOK, 512, 2048, 0);
}
